// FIDLike_5231270166770
// MI455X (gfx1250) — matrix operand order repaired and checked by reading, not compiled, not run on hardware
//
#include <hip/hip_runtime.h>
#include <hip/hip_bf16.h>

#define D 1024
#define NSAMP 32768
#define EPSV 1e-6f
#define LDK 40   // padded k-stride (ushorts) -> 80B rows, 16B aligned chunks

typedef __attribute__((ext_vector_type(16))) __bf16 v16bf;
typedef __attribute__((ext_vector_type(8)))  float  v8f;

union AB16 { uint4 q[2]; v16bf v; };

__device__ __forceinline__ unsigned short f2bf(float f) {
    unsigned int u = __float_as_uint(f);
    u += 0x7FFFu + ((u >> 16) & 1u);   // round-to-nearest-even
    return (unsigned short)(u >> 16);
}

// ---- shared WMMA tile compute: 2x4 fragments (32x64 region) per wave ----
__device__ __forceinline__ void mma_tile(const unsigned short* As,
                                         const unsigned short* Bs,
                                         int wm0, int wn0, int lane,
                                         v8f acc[2][4]) {
    AB16 af[2], bfrag[4];
    const int l15 = lane & 15;
    const int ka = (lane < 16) ? 0 : 8;    // A: lanes 0-15 K0..7,16..23 ; 16-31 K8..15,24..31
    #pragma unroll
    for (int fm = 0; fm < 2; fm++) {
        int m = wm0 + fm * 16 + l15;
        af[fm].q[0] = *(const uint4*)&As[m * LDK + ka];
        af[fm].q[1] = *(const uint4*)&As[m * LDK + ka + 16];
    }
    const int kb = (lane < 16) ? 0 : 8;    // B: lanes 0-15 K0..7,16..23 ; 16-31 K8..15,24..31
    #pragma unroll
    for (int fn = 0; fn < 4; fn++) {
        int n = wn0 + fn * 16 + l15;
        bfrag[fn].q[0] = *(const uint4*)&Bs[n * LDK + kb];
        bfrag[fn].q[1] = *(const uint4*)&Bs[n * LDK + kb + 16];
    }
    #pragma unroll
    for (int fm = 0; fm < 2; fm++)
        #pragma unroll
        for (int fn = 0; fn < 4; fn++)
            acc[fm][fn] = __builtin_amdgcn_wmma_f32_16x16x32_bf16(
                false, af[fm].v, false, bfrag[fn].v,
                (short)0, acc[fm][fn], false, false);
}

// ---- Gram kernel: G = X^T X, X is (NSAMP, D) f32. gridDim.z picks matrix ----
__global__ void __launch_bounds__(256) gram_x(const float* __restrict__ Xr,
                                              const float* __restrict__ Xg,
                                              float* __restrict__ Gr,
                                              float* __restrict__ Gg) {
    __shared__ __align__(16) unsigned short As[128 * LDK];
    __shared__ __align__(16) unsigned short Bs[128 * LDK];
    const float* X = blockIdx.z ? Xg : Xr;
    float*       G = blockIdx.z ? Gg : Gr;

    const int tid  = threadIdx.x;
    const int lane = tid & 31;
    const int wid  = tid >> 5;
    const int wm0  = (wid >> 1) * 32;
    const int wn0  = (wid & 1) * 64;
    const int bm0  = blockIdx.y * 128;
    const int bn0  = blockIdx.x * 128;

    const v8f vzero = {0.f,0.f,0.f,0.f,0.f,0.f,0.f,0.f};
    v8f acc[2][4];
    for (int i = 0; i < 2; i++) for (int j = 0; j < 4; j++) acc[i][j] = vzero;

    float4 ra[4], rb[4];
    auto loadg = [&](int k0) {
        #pragma unroll
        for (int i = 0; i < 4; i++) {
            int idx = tid + i * 256;
            int r = idx >> 5, cg = idx & 31;           // r: k-row 0..31
            ra[i] = *(const float4*)(X + (size_t)(k0 + r) * D + bm0 + cg * 4);
            rb[i] = *(const float4*)(X + (size_t)(k0 + r) * D + bn0 + cg * 4);
        }
    };
    auto stage = [&]() {                                // transpose-store -> [col][k]
        #pragma unroll
        for (int i = 0; i < 4; i++) {
            int idx = tid + i * 256;
            int r = idx >> 5, cg = idx & 31;
            As[(cg*4+0)*LDK + r] = f2bf(ra[i].x);
            As[(cg*4+1)*LDK + r] = f2bf(ra[i].y);
            As[(cg*4+2)*LDK + r] = f2bf(ra[i].z);
            As[(cg*4+3)*LDK + r] = f2bf(ra[i].w);
            Bs[(cg*4+0)*LDK + r] = f2bf(rb[i].x);
            Bs[(cg*4+1)*LDK + r] = f2bf(rb[i].y);
            Bs[(cg*4+2)*LDK + r] = f2bf(rb[i].z);
            Bs[(cg*4+3)*LDK + r] = f2bf(rb[i].w);
        }
    };

    loadg(0);
    const int nk = NSAMP / 32;
    for (int kk = 0; kk < nk; ++kk) {
        __syncthreads();
        stage();
        __syncthreads();
        if (kk + 1 < nk) loadg((kk + 1) * 32);          // overlap VMEM with WMMA
        mma_tile(As, Bs, wm0, wn0, lane, acc);
    }

    #pragma unroll
    for (int fm = 0; fm < 2; fm++)
        #pragma unroll
        for (int fn = 0; fn < 4; fn++) {
            int col   = bn0 + wn0 + fn * 16 + (lane & 15);
            int rbase = bm0 + wm0 + fm * 16 + ((lane < 16) ? 0 : 8);
            #pragma unroll
            for (int r = 0; r < 8; r++)
                G[(size_t)(rbase + r) * D + col] = acc[fm][fn][r];
        }
}

// ---- generic NS GEMM: C = alpha*(A@B) + gamma*I, all DxD row-major f32 ----
__global__ void __launch_bounds__(256) gemm_ns(float* __restrict__ Cout,
                                               const float* __restrict__ A,
                                               const float* __restrict__ B,
                                               float alpha, float gamma) {
    __shared__ __align__(16) unsigned short As[128 * LDK];
    __shared__ __align__(16) unsigned short Bs[128 * LDK];
    const int tid  = threadIdx.x;
    const int lane = tid & 31;
    const int wid  = tid >> 5;
    const int wm0  = (wid >> 1) * 32;
    const int wn0  = (wid & 1) * 64;
    const int bm0  = blockIdx.y * 128;
    const int bn0  = blockIdx.x * 128;

    const v8f vzero = {0.f,0.f,0.f,0.f,0.f,0.f,0.f,0.f};
    v8f acc[2][4];
    for (int i = 0; i < 2; i++) for (int j = 0; j < 4; j++) acc[i][j] = vzero;

    float4 ra[4], rb[4];
    auto loadg = [&](int k0) {
        #pragma unroll
        for (int i = 0; i < 4; i++) {
            int idx = tid + i * 256;
            int m = idx >> 3, kg = idx & 7;            // A natural [m][k]
            ra[i] = *(const float4*)(A + (size_t)(bm0 + m) * D + k0 + kg * 4);
            int r = idx >> 5, cg = idx & 31;           // B [k][n] -> transpose
            rb[i] = *(const float4*)(B + (size_t)(k0 + r) * D + bn0 + cg * 4);
        }
    };
    auto stage = [&]() {
        #pragma unroll
        for (int i = 0; i < 4; i++) {
            int idx = tid + i * 256;
            int m = idx >> 3, kg = idx & 7;
            unsigned short* p = &As[m * LDK + kg * 4];
            p[0] = f2bf(ra[i].x); p[1] = f2bf(ra[i].y);
            p[2] = f2bf(ra[i].z); p[3] = f2bf(ra[i].w);
            int r = idx >> 5, cg = idx & 31;
            Bs[(cg*4+0)*LDK + r] = f2bf(rb[i].x);
            Bs[(cg*4+1)*LDK + r] = f2bf(rb[i].y);
            Bs[(cg*4+2)*LDK + r] = f2bf(rb[i].z);
            Bs[(cg*4+3)*LDK + r] = f2bf(rb[i].w);
        }
    };

    loadg(0);
    const int nk = D / 32;
    for (int kk = 0; kk < nk; ++kk) {
        __syncthreads();
        stage();
        __syncthreads();
        if (kk + 1 < nk) loadg((kk + 1) * 32);
        mma_tile(As, Bs, wm0, wn0, lane, acc);
    }

    #pragma unroll
    for (int fm = 0; fm < 2; fm++)
        #pragma unroll
        for (int fn = 0; fn < 4; fn++) {
            int col   = bn0 + wn0 + fn * 16 + (lane & 15);
            int rbase = bm0 + wm0 + fm * 16 + ((lane < 16) ? 0 : 8);
            #pragma unroll
            for (int r = 0; r < 8; r++) {
                int row = rbase + r;
                float v = alpha * acc[fm][fn][r];
                if (row == col) v += gamma;
                Cout[(size_t)row * D + col] = v;
            }
        }
}

// ---- small support kernels (deterministic tree reductions, no atomics) ----
__global__ void colsum_partial(const float* __restrict__ X, float* __restrict__ part) {
    int c  = blockIdx.x * 256 + threadIdx.x;   // gridDim.x = 4
    int rb = blockIdx.y;                       // gridDim.y = 32
    float s = 0.f;
    for (int r = rb * 1024; r < (rb + 1) * 1024; ++r) s += X[(size_t)r * D + c];
    part[rb * D + c] = s;
}

__global__ void colsum_finish(const float* __restrict__ part, float* __restrict__ mu) {
    int c = blockIdx.x * 256 + threadIdx.x;
    float s = 0.f;
    for (int rb = 0; rb < 32; ++rb) s += part[rb * D + c];
    mu[c] = s / (float)NSAMP;
}

__global__ void diffmu_kernel(const float* __restrict__ a, const float* __restrict__ b,
                              float* __restrict__ out) {
    __shared__ float red[256];
    int tid = threadIdx.x;
    float s = 0.f;
    for (int i = tid; i < D; i += 256) { float d = a[i] - b[i]; s += d * d; }
    red[tid] = s; __syncthreads();
    for (int off = 128; off > 0; off >>= 1) {
        if (tid < off) red[tid] += red[tid + off];
        __syncthreads();
    }
    if (tid == 0) *out = red[0];
}

__global__ void cov_finalize(float* __restrict__ G, const float* __restrict__ mu) {
    int i = blockIdx.x * 256 + threadIdx.x;    // grid 4096
    int r = i >> 10, c = i & 1023;
    float v = (G[i] - (float)NSAMP * mu[r] * mu[c]) * (1.f / (float)(NSAMP - 1));
    if (r == c) v += EPSV;
    G[i] = v;
}

__global__ void trace_diag(const float* __restrict__ A, float* __restrict__ out) {
    __shared__ float red[256];
    int tid = threadIdx.x;
    float s = 0.f;
    for (int i = tid; i < D; i += 256) s += A[(size_t)i * D + i];
    red[tid] = s; __syncthreads();
    for (int off = 128; off > 0; off >>= 1) {
        if (tid < off) red[tid] += red[tid + off];
        __syncthreads();
    }
    if (tid == 0) *out = red[0];
}

__global__ void sumsq_partial(const float* __restrict__ A, float* __restrict__ part) {
    __shared__ float red[256];
    int tid = threadIdx.x;
    float s = 0.f;
    for (int i = blockIdx.x * 256 + tid; i < D * D; i += 256 * 256) {
        float v = A[i]; s += v * v;
    }
    red[tid] = s; __syncthreads();
    for (int off = 128; off > 0; off >>= 1) {
        if (tid < off) red[tid] += red[tid + off];
        __syncthreads();
    }
    if (tid == 0) part[blockIdx.x] = red[0];
}

__global__ void finalize_norm(const float* __restrict__ part, float* __restrict__ c) {
    __shared__ float red[256];
    int tid = threadIdx.x;
    red[tid] = part[tid]; __syncthreads();
    for (int off = 128; off > 0; off >>= 1) {
        if (tid < off) red[tid] += red[tid + off];
        __syncthreads();
    }
    if (tid == 0) *c = sqrtf(red[0]);          // Frobenius norm >= spectral norm
}

__global__ void ns_init(const float* __restrict__ A, const float* __restrict__ c,
                        float* __restrict__ Y, float* __restrict__ Z) {
    int i = blockIdx.x * 256 + threadIdx.x;
    float inv = 1.f / *c;
    Y[i] = A[i] * inv;
    int r = i >> 10, cc = i & 1023;
    Z[i] = (r == cc) ? 1.f : 0.f;
}

__global__ void scale_sqrtc(const float* __restrict__ Y, const float* __restrict__ c,
                            float* __restrict__ S) {
    int i = blockIdx.x * 256 + threadIdx.x;
    S[i] = Y[i] * sqrtf(*c);
}

__global__ void final_combine(const float* __restrict__ scal, float* __restrict__ out) {
    // scal: [1]=c_m, [2]=diff_mu, [3]=tr_r, [4]=tr_g, [5]=tr(Y_M)
    out[0] = scal[2] + scal[3] + scal[4] - 2.f * sqrtf(scal[1]) * scal[5];
}

extern "C" void kernel_launch(void* const* d_in, const int* in_sizes, int n_in,
                              void* d_out, int out_size, void* d_ws, size_t ws_size,
                              hipStream_t stream) {
    (void)in_sizes; (void)n_in; (void)out_size; (void)ws_size;
    const float* Xr = (const float*)d_in[0];
    const float* Xg = (const float*)d_in[1];
    float* out = (float*)d_out;
    float* w   = (float*)d_ws;

    const size_t MM = (size_t)D * D;
    float* scal   = w;                 // 16 scalars
    float* part   = w + 16;            // 256 sumsq partials
    float* mu_r   = w + 512;           // D
    float* mu_g   = w + 1536;          // D
    float* muPart = w + 2560;          // 32*D
    float* base   = w + 40960;
    float* covR = base;          float* covG = covR + MM;
    float* Y0 = covG + MM;       float* Y1 = Y0 + MM;
    float* Z0 = Y1 + MM;         float* Z1 = Z0 + MM;
    float* T  = Z1 + MM;         float* S  = T  + MM;
    float* Mm = S + MM;          float* tmp = Mm + MM;

    dim3 g88(8, 8), g882(8, 8, 2);

    // means + mu-diff
    colsum_partial<<<dim3(4, 32), 256, 0, stream>>>(Xr, muPart);
    colsum_finish<<<4, 256, 0, stream>>>(muPart, mu_r);
    colsum_partial<<<dim3(4, 32), 256, 0, stream>>>(Xg, muPart);
    colsum_finish<<<4, 256, 0, stream>>>(muPart, mu_g);
    diffmu_kernel<<<1, 256, 0, stream>>>(mu_r, mu_g, &scal[2]);

    // Gram matrices via WMMA, then covariance correction
    gram_x<<<g882, 256, 0, stream>>>(Xr, Xg, covR, covG);
    cov_finalize<<<4096, 256, 0, stream>>>(covR, mu_r);
    cov_finalize<<<4096, 256, 0, stream>>>(covG, mu_g);
    trace_diag<<<1, 256, 0, stream>>>(covR, &scal[3]);
    trace_diag<<<1, 256, 0, stream>>>(covG, &scal[4]);

    // Newton-Schulz sqrtm(covR) -> S
    sumsq_partial<<<256, 256, 0, stream>>>(covR, part);
    finalize_norm<<<1, 256, 0, stream>>>(part, &scal[0]);
    ns_init<<<4096, 256, 0, stream>>>(covR, &scal[0], Y0, Z0);
    {
        float *Y = Y0, *Z = Z0, *Yn = Y1, *Zn = Z1;
        for (int it = 0; it < 12; ++it) {
            gemm_ns<<<g88, 256, 0, stream>>>(T,  Z, Y, -0.5f, 1.5f);  // T = 1.5I - 0.5 ZY
            gemm_ns<<<g88, 256, 0, stream>>>(Yn, Y, T,  1.0f, 0.0f);  // Y' = Y T
            gemm_ns<<<g88, 256, 0, stream>>>(Zn, T, Z,  1.0f, 0.0f);  // Z' = T Z
            float* t0 = Y; Y = Yn; Yn = t0;
            float* t1 = Z; Z = Zn; Zn = t1;
        }
        scale_sqrtc<<<4096, 256, 0, stream>>>(Y, &scal[0], S);        // S = sqrt(c)*Y
    }

    // M = S @ covG @ S
    gemm_ns<<<g88, 256, 0, stream>>>(tmp, S, covG, 1.0f, 0.0f);
    gemm_ns<<<g88, 256, 0, stream>>>(Mm, tmp, S, 1.0f, 0.0f);

    // Newton-Schulz sqrtm(M): trace(sqrtm(M)) = sqrt(c_m) * trace(Y_final)
    sumsq_partial<<<256, 256, 0, stream>>>(Mm, part);
    finalize_norm<<<1, 256, 0, stream>>>(part, &scal[1]);
    ns_init<<<4096, 256, 0, stream>>>(Mm, &scal[1], Y0, Z0);
    {
        float *Y = Y0, *Z = Z0, *Yn = Y1, *Zn = Z1;
        for (int it = 0; it < 12; ++it) {
            gemm_ns<<<g88, 256, 0, stream>>>(T,  Z, Y, -0.5f, 1.5f);
            gemm_ns<<<g88, 256, 0, stream>>>(Yn, Y, T,  1.0f, 0.0f);
            gemm_ns<<<g88, 256, 0, stream>>>(Zn, T, Z,  1.0f, 0.0f);
            float* t0 = Y; Y = Yn; Yn = t0;
            float* t1 = Z; Z = Zn; Zn = t1;
        }
        trace_diag<<<1, 256, 0, stream>>>(Y, &scal[5]);
    }

    final_combine<<<1, 1, 0, stream>>>(scal, out);
}
